// LKA_dynamic_80968723464254
// MI455X (gfx1250) — matrix operand order repaired and checked by reading, not compiled, not run on hardware
//
#include <hip/hip_runtime.h>
#include <math.h>

#define B_ 8
#define C_ 256
#define H_ 128
#define W_ 128
#define HW_ (H_*W_)
#define E_ 3

typedef __attribute__((ext_vector_type(16))) _Float16 v16h;
typedef __attribute__((ext_vector_type(8)))  _Float16 v8h;
typedef __attribute__((ext_vector_type(8)))  float    v8f;

// 128-bit vector type matching the async-LDS builtin's parameter type
typedef int v4i_ __attribute__((vector_size(4 * sizeof(int))));
typedef __attribute__((address_space(1))) v4i_* gv4i_p;  // global (AS1) int4*
typedef __attribute__((address_space(3))) v4i_* lv4i_p;  // LDS (AS3) int4*

#if defined(__has_builtin)
#  if __has_builtin(__builtin_amdgcn_global_load_async_to_lds_b128)
#    define HAVE_ASYNC_LDS 1
#  endif
#endif
#ifndef HAVE_ASYNC_LDS
#  define HAVE_ASYNC_LDS 0
#endif

__device__ __forceinline__ float gelu_exact(float v) {
    return 0.5f * v * (1.0f + erff(v * 0.70710678118654752f));
}

// ---------------------------------------------------------------------------
// pooled[b*C+c] = mean over the HxW plane (one block per (b,c) plane)
// ---------------------------------------------------------------------------
__global__ void pool_kernel(const float* __restrict__ in, float* __restrict__ pooled) {
    int bc = blockIdx.x;
    int t  = threadIdx.x;
    const float* p = in + (size_t)bc * HW_;
    float s = 0.f;
    for (int i = t; i < HW_; i += 256) s += p[i];
    __shared__ float red[256];
    red[t] = s;
    __syncthreads();
    for (int k = 128; k > 0; k >>= 1) {
        if (t < k) red[t] += red[t + k];
        __syncthreads();
    }
    if (t == 0) pooled[bc] = red[0] * (1.0f / (float)HW_);
}

// ---------------------------------------------------------------------------
// routing MLP + expert mix. One block (256 threads) per batch sample.
// ---------------------------------------------------------------------------
__global__ void route_mix_kernel(const float* __restrict__ pooled,
                                 const float* __restrict__ fc1w,  // [C,C]
                                 const float* __restrict__ fc1b,  // [C]
                                 const float* __restrict__ fc2w,  // [C,E]
                                 const float* __restrict__ fc2b,  // [E]
                                 const float* __restrict__ experts, // [E,C,KK]
                                 float* __restrict__ mk, int KK)     // [B,C,KK]
{
    int b = blockIdx.x, t = threadIdx.x;
    __shared__ float hbuf[C_];
    __shared__ float rbuf[E_];
    const float* pb = pooled + b * C_;
    float s = fc1b[t];
    for (int c = 0; c < C_; ++c) s += pb[c] * fc1w[c * C_ + t];
    hbuf[t] = (s >= 0.f) ? s : 0.1f * s;
    __syncthreads();
    if (t < E_) {
        float a = fc2b[t];
        for (int j = 0; j < C_; ++j) a += hbuf[j] * fc2w[j * E_ + t];
        rbuf[t] = 1.0f / (1.0f + __expf(-a));
    }
    __syncthreads();
    float r0 = rbuf[0], r1 = rbuf[1], r2 = rbuf[2];
    int n = C_ * KK;
    for (int i = t; i < n; i += 256)
        mk[b * n + i] = r0 * experts[i] + r1 * experts[n + i] + r2 * experts[2 * n + i];
}

// ---------------------------------------------------------------------------
// fp32 -> f16 convert (for w1, so the GEMM A tile becomes a byte-exact copy)
// ---------------------------------------------------------------------------
__global__ void f32_to_f16_kernel(const float* __restrict__ in,
                                  _Float16* __restrict__ out, int n) {
    int i = blockIdx.x * 256 + threadIdx.x;
    if (i < n) out[i] = (_Float16)in[i];
}

// ---------------------------------------------------------------------------
// Depthwise conv (per-sample mixed kernel) + bias + exact GELU.
// One block per (b,c, row-tile). Full-width LDS row tile with vertical halo.
// OT = output element type (float for stage 1, _Float16 for stage 2 so the
// WMMA GEMM reads half the bytes from HBM).
// ---------------------------------------------------------------------------
template <int KS, int DIL, typename OT>
__global__ void dwconv_gelu_kernel(const float* __restrict__ in,
                                   const float* __restrict__ mk,   // [B*C, KS*KS]
                                   const float* __restrict__ bias, // [C]
                                   OT* __restrict__ out)
{
    constexpr int HALO = (KS / 2) * DIL;
    constexpr int TR   = 32;
    constexpr int LR   = TR + 2 * HALO;
    __shared__ float tile[LR * W_];
    __shared__ float kw[KS * KS];

    int bc = blockIdx.x;
    int c  = bc % C_;
    int r0 = blockIdx.y * TR;
    int t  = threadIdx.x;
    const float* plane = in + (size_t)bc * HW_;

    for (int i = t; i < LR * W_; i += 256) {
        int lr = i / W_, col = i % W_;
        int gr = r0 - HALO + lr;
        tile[i] = (gr >= 0 && gr < H_) ? plane[gr * W_ + col] : 0.f;
    }
    if (t < KS * KS) kw[t] = mk[bc * (KS * KS) + t];
    __syncthreads();

    float bval = bias[c];
    OT* oplane = out + (size_t)bc * HW_;
    for (int i = t; i < TR * W_; i += 256) {
        int rl = i / W_, col = i % W_;
        float acc = 0.f;
#pragma unroll
        for (int ky = 0; ky < KS; ++ky) {
            int lr = rl + ky * DIL;
#pragma unroll
            for (int kx = 0; kx < KS; ++kx) {
                int gc = col + (kx - KS / 2) * DIL;
                float v = (gc >= 0 && gc < W_) ? tile[lr * W_ + gc] : 0.f;
                acc += kw[ky * KS + kx] * v;
            }
        }
        oplane[(r0 + rl) * W_ + col] = (OT)gelu_exact(acc + bval);
    }
}

// ---------------------------------------------------------------------------
// Pointwise 1x1 conv as WMMA GEMM (f16 in, f32 acc), fused with +b1 and the
// final gate: out = x * (w1 @ a2 + b1).
// WG tile M=128, N=64, K-step 32.  8 waves = 2(M) x 4(N); wave = 64x16 tile.
// A tile (w1h) is staged with GLOBAL_LOAD_ASYNC_TO_LDS_B128 (byte-exact copy,
// tracked by ASYNCcnt); B tile (a2h) is repacked/transposed manually.
// ---------------------------------------------------------------------------
__global__ void pw_wmma_kernel(const _Float16* __restrict__ a2h, // [B,C,HW] f16
                               const _Float16* __restrict__ w1h, // [O=C, C] f16
                               const float* __restrict__ b1,     // [C]
                               const float* __restrict__ x,      // [B,C,HW]
                               float* __restrict__ out)          // [B,C,HW]
{
    constexpr int WGM = 128, WGN = 64, KT = 32, LDA = KT + 8; // row = 80B, chunks 16B aligned
    __shared__ __align__(16) _Float16 Ash[WGM * LDA]; // w1 tile [m][k]
    __shared__ __align__(16) _Float16 Bsh[WGN * LDA]; // a2 tile [n][k] (transposed)

    int n0 = blockIdx.x * WGN;
    int m0 = blockIdx.y * WGM;
    int b  = blockIdx.z;
    int t  = threadIdx.x;
    int lane = t & 31, wave = t >> 5;
    int wm = wave >> 2, wn = wave & 3;
    int g  = lane >> 4, ln = lane & 15;

    const _Float16* Bglob = a2h + (size_t)b * C_ * HW_;

    v8f acc[4];
#pragma unroll
    for (int i = 0; i < 4; ++i) acc[i] = (v8f){};

    for (int k0 = 0; k0 < C_; k0 += KT) {
        // ---- A tile: w1h[m0+m][k0 .. k0+31] -> Ash[m][k] ----
#if HAVE_ASYNC_LDS
        // 128 rows x 64B = 512 chunks of 16B; ASYNCcnt-tracked, no VGPR staging
#pragma unroll
        for (int i = 0; i < 2; ++i) {
            int cid = i * 256 + t;
            int m = cid >> 2, q = cid & 3;
            const _Float16* src = w1h + (size_t)(m0 + m) * C_ + k0 + q * 8;
            _Float16* dst = &Ash[m * LDA + q * 8];
            __builtin_amdgcn_global_load_async_to_lds_b128(
                (gv4i_p)(const void*)src, (lv4i_p)(void*)dst, 0, 0);
        }
#else
#pragma unroll
        for (int i = 0; i < (WGM * KT) / 256; ++i) {
            int idx = i * 256 + t;
            int m = idx >> 5, k = idx & 31;
            Ash[m * LDA + k] = w1h[(size_t)(m0 + m) * C_ + (k0 + k)];
        }
#endif
        // ---- B tile: a2h[k0+k][n0..n0+63] -> Bsh[n][k] (transpose repack) ----
#pragma unroll
        for (int i = 0; i < 4; ++i) {
            int idx = i * 256 + t;
            int j = idx & 31, k = idx >> 5;     // j = half-pair column, k = row
            union { unsigned u; _Float16 h[2]; } cv;
            cv.u = *(const unsigned*)&Bglob[(size_t)(k0 + k) * HW_ + n0 + 2 * j];
            Bsh[(2 * j) * LDA + k]     = cv.h[0];
            Bsh[(2 * j + 1) * LDA + k] = cv.h[1];
        }
        // prefetch next a2 K-tile (global_prefetch_b8)
        if (k0 + KT < C_) {
            const _Float16* nb = Bglob + (size_t)(k0 + KT + (t >> 6)) * HW_ + n0 + 2 * (t & 63);
            __builtin_prefetch(nb, 0, 1);
        }
#if HAVE_ASYNC_LDS
#  if __has_builtin(__builtin_amdgcn_s_wait_asynccnt)
        __builtin_amdgcn_s_wait_asynccnt(0);
#  else
        asm volatile("s_wait_asynccnt 0" ::: "memory");
#  endif
#endif
        __syncthreads();

        // B fragment: lane covers K in {8g..8g+7} U {16+8g..16+8g+7} at column n (matches A)
        int nn = wn * 16 + ln;
        v8h blo = *(const v8h*)&Bsh[nn * LDA + 8 * g];
        v8h bhi = *(const v8h*)&Bsh[nn * LDA + 16 + 8 * g];
        v16h bf;
#pragma unroll
        for (int i = 0; i < 8; ++i) { bf[i] = blo[i]; bf[i + 8] = bhi[i]; }

        // A fragments: lane covers K in {8g..8g+7} U {16+8g..16+8g+7} (ISA A 16x32 layout)
#pragma unroll
        for (int ts = 0; ts < 4; ++ts) {
            int row = wm * 64 + ts * 16 + ln;
            v8h a0 = *(const v8h*)&Ash[row * LDA + 8 * g];
            v8h a1 = *(const v8h*)&Ash[row * LDA + 16 + 8 * g];
            v16h af;
#pragma unroll
            for (int i = 0; i < 8; ++i) { af[i] = a0[i]; af[i + 8] = a1[i]; }
            acc[ts] = __builtin_amdgcn_wmma_f32_16x16x32_f16(
                false, af, false, bf, (short)0, acc[ts], false, false);
        }
        __syncthreads();
    }

    // epilogue: + b1[o], gate with x, write out  (C/D layout: M = r + 8g, N = ln)
    int p = n0 + wn * 16 + ln;
    size_t base = (size_t)b * C_ * HW_;
#pragma unroll
    for (int ts = 0; ts < 4; ++ts) {
#pragma unroll
        for (int r = 0; r < 8; ++r) {
            int m = m0 + wm * 64 + ts * 16 + r + 8 * g;
            size_t off = base + (size_t)m * HW_ + p;
            float val = acc[ts][r] + b1[m];
            out[off] = val * x[off];
        }
    }
}

// ---------------------------------------------------------------------------
extern "C" void kernel_launch(void* const* d_in, const int* in_sizes, int n_in,
                              void* d_out, int out_size, void* d_ws, size_t ws_size,
                              hipStream_t stream)
{
    (void)in_sizes; (void)n_in; (void)out_size; (void)ws_size;
    const float* x     = (const float*)d_in[0];
    const float* w0e   = (const float*)d_in[1];   // [3,256,5,5]
    const float* b0    = (const float*)d_in[2];
    const float* fc1w0 = (const float*)d_in[3];
    const float* fc1b0 = (const float*)d_in[4];
    const float* fc2w0 = (const float*)d_in[5];
    const float* fc2b0 = (const float*)d_in[6];
    const float* wse   = (const float*)d_in[7];   // [3,256,7,7]
    const float* bs    = (const float*)d_in[8];
    const float* fc1ws = (const float*)d_in[9];
    const float* fc1bs = (const float*)d_in[10];
    const float* fc2ws = (const float*)d_in[11];
    const float* fc2bs = (const float*)d_in[12];
    const float* w1    = (const float*)d_in[13];  // [O,C]
    const float* b1    = (const float*)d_in[14];
    float* out = (float*)d_out;

    // workspace layout (bytes): a2h (64MB, f16) | w1h (128KB, f16) | pooled | mk0 | mks
    char* wsb = (char*)d_ws;
    _Float16* a2h  = (_Float16*)wsb;
    _Float16* w1h  = (_Float16*)(wsb + (size_t)B_ * C_ * HW_ * sizeof(_Float16));
    float* pooled  = (float*)(wsb + (size_t)B_ * C_ * HW_ * sizeof(_Float16)
                                   + (size_t)C_ * C_ * sizeof(_Float16));
    float* mk0 = pooled + B_ * C_;       // B*C*25
    float* mks = mk0 + B_ * C_ * 25;     // B*C*49
    float* a1  = out;  // stage-1 activation staged in d_out (consumed before final write)

    // ---- stage 1: 5x5 depthwise cond-conv + GELU ----
    pool_kernel<<<dim3(B_ * C_), 256, 0, stream>>>(x, pooled);
    route_mix_kernel<<<dim3(B_), 256, 0, stream>>>(pooled, fc1w0, fc1b0, fc2w0, fc2b0, w0e, mk0, 25);
    dwconv_gelu_kernel<5, 1, float><<<dim3(B_ * C_, H_ / 32), 256, 0, stream>>>(x, mk0, b0, a1);

    // ---- stage 2: 7x7 dil-3 depthwise cond-conv + GELU (f16 output) ----
    pool_kernel<<<dim3(B_ * C_), 256, 0, stream>>>(a1, pooled);
    route_mix_kernel<<<dim3(B_), 256, 0, stream>>>(pooled, fc1ws, fc1bs, fc2ws, fc2bs, wse, mks, 49);
    dwconv_gelu_kernel<7, 3, _Float16><<<dim3(B_ * C_, H_ / 32), 256, 0, stream>>>(a1, mks, bs, a2h);

    // ---- stage 3: WMMA pointwise conv fused with bias + gate ----
    f32_to_f16_kernel<<<dim3((C_ * C_ + 255) / 256), 256, 0, stream>>>(w1, w1h, C_ * C_);
    pw_wmma_kernel<<<dim3(HW_ / 64, C_ / 128, B_), 256, 0, stream>>>(a2h, w1h, b1, x, out);
}
